// UnifiedGNNLayer_49606872269034
// MI455X (gfx1250) — compile-verified, matrix operand order checked by reading, not run on hardware
//
#include <hip/hip_runtime.h>

#define DD 128
#define NHEAD 8
#define MAXSEQ 50

typedef __attribute__((ext_vector_type(16))) __bf16          v16bf;
typedef __attribute__((ext_vector_type(16))) unsigned short  v16us;
typedef __attribute__((ext_vector_type(8)))  unsigned short  u16x8;
typedef __attribute__((ext_vector_type(8)))  float           v8f;
typedef __attribute__((ext_vector_type(4)))  int             i32x4;

#if defined(__has_builtin)
#  if __has_builtin(__builtin_amdgcn_global_load_async_to_lds_b128)
#    define ASYNC_LDS_COPY 1
#  endif
#endif
#ifndef ASYNC_LDS_COPY
#  define ASYNC_LDS_COPY 0
#endif

// ---------- small device helpers ----------
__device__ __forceinline__ unsigned short f2bf(float f) {
  unsigned u = __builtin_bit_cast(unsigned, f);
  u += 0x7FFFu + ((u >> 16) & 1u);          // round-to-nearest-even
  return (unsigned short)(u >> 16);
}
__device__ __forceinline__ unsigned mapf(float f) {   // order-preserving float->uint
  unsigned u = __builtin_bit_cast(unsigned, f);
  return (u & 0x80000000u) ? ~u : (u | 0x80000000u);
}
__device__ __forceinline__ float unmapf(unsigned m) {
  unsigned u = (m & 0x80000000u) ? (m ^ 0x80000000u) : ~m;
  return __builtin_bit_cast(float, u);
}

// Build a 16-element bf16 WMMA fragment from two 16-byte chunks.
// Per ISA 16-bit layout: lanes 0-15 hold K = k..k+7 and k+16..k+23,
// lanes 16-31 hold K = k+8..k+15 and k+24..k+31  (khalf = (lane>>4)*8).
__device__ __forceinline__ v16bf make_frag(const unsigned short* p, int khalf) {
  u16x8 lo = *(const u16x8*)(p + khalf);
  u16x8 hi = *(const u16x8*)(p + 16 + khalf);
  v16us v;
#pragma unroll
  for (int j = 0; j < 8; ++j) { v[j] = lo[j]; v[j + 8] = hi[j]; }
  return __builtin_bit_cast(v16bf, v);
}

// ---------- generic WMMA GEMM:  C[M,128] = A[M,K] (bf16) x Bt[128,K]^T (bf16) ----------
// MODE 0: A0 dense [M,K]
// MODE 1: row m = concat( A0[idx0[m]][0:128], A0[idx1[m]][0:128] )   (K==256, edge gather)
// MODE 2: row m = concat( A0[m], A1[m], A2[m] )                      (K==384, fusion)
// Block = 256 threads = 8 waves. Block tile = 32 rows x 128 cols.
// A tile (32 x K bf16) is staged once into LDS via async global->LDS copy (or
// load+ds_store fallback), then each wave computes TWO 16x16 tiles sharing one
// B fragment. Epilogue: +bias, ReLU, +residual(f32), atomic scatter-add by
// scatIdx, row type mask, optional f32 and/or bf16 stores.
template <int MODE, int K>
__global__ __launch_bounds__(256) void wmma_gemm_kernel(
    const unsigned short* __restrict__ A0, const unsigned short* __restrict__ A1,
    const unsigned short* __restrict__ A2,
    const int* __restrict__ idx0, const int* __restrict__ idx1,
    const unsigned short* __restrict__ Bt,
    const float* __restrict__ bias, const float* __restrict__ resid,
    float* __restrict__ outF, unsigned short* __restrict__ outB,
    float* __restrict__ scat, const int* __restrict__ scatIdx,
    const int* __restrict__ typeArr, int typeVal,
    int M, int doRelu)
{
  __shared__ __align__(16) unsigned short smA[32 * K];

  const int lane  = threadIdx.x & 31;
  const int wave  = threadIdx.x >> 5;          // 8 waves -> 8 column tiles (Nc = 128)
  const int m0    = blockIdx.x * 32;
  const int lm    = lane & 15;
  const int khalf = (lane >> 4) * 8;
  const int n     = wave * 16 + lm;

  // ---- stage A block-tile (32 rows x K bf16 = 32*K*2 bytes) into LDS ----
  // 16-byte chunks, 256 threads -> K/64 chunks per thread.
#pragma unroll
  for (int i = 0; i < K / 64; ++i) {
    const int c    = threadIdx.x + 256 * i;
    const int boff = c * 16;                   // byte offset within LDS tile
    const int rowi = boff / (K * 2);
    const int rb   = boff - rowi * (K * 2);    // byte offset within row
    int mr = m0 + rowi;
    if (mr >= M) mr = M - 1;                   // clamp; stores are guarded later
    const char* src;
    if constexpr (MODE == 0) {
      src = (const char*)A0 + (size_t)mr * (K * 2) + rb;
    } else if constexpr (MODE == 1) {
      src = (rb < 256) ? (const char*)A0 + (size_t)idx0[mr] * 256 + rb
                       : (const char*)A0 + (size_t)idx1[mr] * 256 + (rb - 256);
    } else {
      src = (rb < 256) ? (const char*)A0 + (size_t)mr * 256 + rb
          : (rb < 512) ? (const char*)A1 + (size_t)mr * 256 + (rb - 256)
                       : (const char*)A2 + (size_t)mr * 256 + (rb - 512);
    }
#if ASYNC_LDS_COPY
    __builtin_amdgcn_global_load_async_to_lds_b128(
        (i32x4*)(void*)const_cast<char*>(src),
        (i32x4*)(void*)((char*)smA + boff), 0, 0);
#else
    u16x8 tmp = *(const u16x8*)src;
    *(u16x8*)((char*)smA + boff) = tmp;
#endif
  }
#if ASYNC_LDS_COPY
#  if __has_builtin(__builtin_amdgcn_s_wait_asynccnt)
  __builtin_amdgcn_s_wait_asynccnt(0);
#  else
  asm volatile("s_wait_asynccnt 0" ::: "memory");
#  endif
#endif
  __syncthreads();

  // ---- compute: two 16x16 tiles per wave sharing one B fragment ----
  v8f acc0 = {}, acc1 = {};
  const unsigned short* bptr = Bt + (size_t)n * K;
  const int a0base = lm * K;                   // rows  0..15 of block tile
  const int a1base = (16 + lm) * K;            // rows 16..31 of block tile

#pragma unroll
  for (int k0 = 0; k0 < K; k0 += 32) {
    __builtin_prefetch(bptr + k0 + 32, 0, 3);  // -> global_prefetch_b8
    v16bf b  = make_frag(bptr + k0, khalf);    // global (L2-resident weights)
    v16bf a0 = make_frag(smA + a0base + k0, khalf);  // ds_load_b128 x2
    v16bf a1 = make_frag(smA + a1base + k0, khalf);
    acc0 = __builtin_amdgcn_wmma_f32_16x16x32_bf16(
        false, a0, false, b, (short)0, acc0, false, false);
    acc1 = __builtin_amdgcn_wmma_f32_16x16x32_bf16(
        false, a1, false, b, (short)0, acc1, false, false);
  }

  // ---- epilogue ----
  const float bn    = bias ? bias[n] : 0.0f;
  const int   rbase = (lane >> 4) << 3;        // lanes 16-31 hold rows +8..+15
#pragma unroll
  for (int t = 0; t < 2; ++t) {
    const v8f acc = t ? acc1 : acc0;
    const int mt  = m0 + t * 16;
#pragma unroll
    for (int r = 0; r < 8; ++r) {
      const int mr = mt + rbase + r;
      if (mr >= M) continue;
      if (typeArr && typeArr[mr] != typeVal) continue;
      float v = acc[r] + bn;
      if (doRelu) v = v > 0.0f ? v : 0.0f;
      if (resid) v += resid[(size_t)mr * DD + n];
      if (scat)  atomicAdd(&scat[(size_t)scatIdx[mr] * DD + n], v);
      if (outF)  outF[(size_t)mr * DD + n] = v;
      if (outB)  outB[(size_t)mr * DD + n] = f2bf(v);
    }
  }
}

// ---------- elementwise / conversion kernels ----------
__global__ void to_bf16_kernel(const float* __restrict__ a, unsigned short* __restrict__ o, int n) {
  int t = blockIdx.x * 256 + threadIdx.x;
  if (t < n) o[t] = f2bf(a[t]);
}
__global__ void add_bf16_kernel(const float* __restrict__ a, const float* __restrict__ b,
                                unsigned short* __restrict__ o, int n) {
  int t = blockIdx.x * 256 + threadIdx.x;
  if (t < n) o[t] = f2bf(a[t] + b[t]);
}
// dst[128,K] (bf16) = transpose of src[K,128] (f32)
__global__ void conv_transpose_kernel(const float* __restrict__ src,
                                      unsigned short* __restrict__ dst, int K) {
  int t = blockIdx.x * 256 + threadIdx.x;
  if (t >= K * DD) return;
  int nn = t / K, k = t - nn * K;
  dst[t] = f2bf(src[(size_t)k * DD + nn]);
}
__global__ void temporal_embed_kernel(const float* __restrict__ x, const float* __restrict__ emb,
                                      const int* __restrict__ tpos,
                                      float* __restrict__ outF, unsigned short* __restrict__ outB,
                                      int total) {
  int t = blockIdx.x * 256 + threadIdx.x;
  if (t >= total) return;
  int nn = t >> 7, d = t & (DD - 1);
  int p = tpos[nn]; p = p < 0 ? 0 : (p > MAXSEQ - 1 ? MAXSEQ - 1 : p);
  float v = x[t] + emb[p * DD + d];
  outF[t] = v;
  outB[t] = f2bf(v);
}

// ---------- GAT attention ----------
__global__ void gat_scores_kernel(const float* __restrict__ Q, const float* __restrict__ Kf,
                                  const int* __restrict__ row, const int* __restrict__ col,
                                  float* __restrict__ scores, unsigned* __restrict__ mxu, int E) {
  int e = blockIdx.x * 256 + threadIdx.x;
  if (e >= E) return;
  const float* q = Q  + (size_t)row[e] * DD;
  const float* k = Kf + (size_t)col[e] * DD;
  int c = col[e];
#pragma unroll
  for (int h = 0; h < NHEAD; ++h) {
    float s = 0.f;
#pragma unroll
    for (int d = 0; d < 16; ++d) s += q[h * 16 + d] * k[h * 16 + d];
    s *= 0.25f;                                  // 1/sqrt(16)
    s = s > 0.f ? s : 0.2f * s;                  // leaky relu
    scores[(size_t)e * NHEAD + h] = s;
    atomicMax(&mxu[c * NHEAD + h], mapf(s));     // segment max
  }
}
__global__ void gat_exp_kernel(const int* __restrict__ col, float* __restrict__ scores,
                               const unsigned* __restrict__ mxu, float* __restrict__ sm, int E) {
  int t = blockIdx.x * 256 + threadIdx.x;
  if (t >= E * NHEAD) return;
  int e = t >> 3, h = t & 7;
  int c = col[e];
  float ex = __expf(scores[t] - unmapf(mxu[c * NHEAD + h]));
  scores[t] = ex;
  atomicAdd(&sm[c * NHEAD + h], ex);
}
__global__ void gat_agg_kernel(const int* __restrict__ row, const int* __restrict__ col,
                               const float* __restrict__ ex, const float* __restrict__ sm,
                               const float* __restrict__ V, float* __restrict__ agg, int E) {
  int t = blockIdx.x * 256 + threadIdx.x;
  if (t >= E * DD) return;
  int e = t >> 7, d = t & (DD - 1), h = d >> 4;
  int c = col[e], r = row[e];
  float p = ex[(size_t)e * NHEAD + h] / sm[c * NHEAD + h];
  atomicAdd(&agg[(size_t)c * DD + d], p * V[(size_t)r * DD + d]);
}

// ---------- LayerNorm: one wave (32 lanes x 4 values) per node ----------
__global__ void layernorm_kernel(const float* __restrict__ in, const float* __restrict__ g,
                                 const float* __restrict__ b, float* __restrict__ outF,
                                 unsigned short* __restrict__ outB, int N) {
  int idx  = blockIdx.x * blockDim.x + threadIdx.x;
  int node = idx >> 5;
  int lane = idx & 31;
  if (node >= N) return;
  const float4 v = ((const float4*)(in + (size_t)node * DD))[lane];
  float s = v.x + v.y + v.z + v.w;
#pragma unroll
  for (int o = 16; o; o >>= 1) s += __shfl_xor(s, o, 32);
  float mean = s * (1.0f / DD);
  float dx = v.x - mean, dy = v.y - mean, dz = v.z - mean, dw = v.w - mean;
  float q = dx * dx + dy * dy + dz * dz + dw * dw;
#pragma unroll
  for (int o = 16; o; o >>= 1) q += __shfl_xor(q, o, 32);
  float inv = rsqrtf(q * (1.0f / DD) + 1e-5f);
  float rr[4] = {dx, dy, dz, dw};
#pragma unroll
  for (int i = 0; i < 4; ++i) {
    int d = lane * 4 + i;
    float o = rr[i] * inv * g[d] + b[d];
    if (outF) outF[(size_t)node * DD + d] = o;
    if (outB) outB[(size_t)node * DD + d] = f2bf(o);
  }
}

// ---------- host-side templated launch helper ----------
template <int MODE, int K>
static void launch_gemm(const void* A0, const void* A1, const void* A2,
                        const int* i0, const int* i1, const void* Bt,
                        const float* bias, const float* resid,
                        float* outF, unsigned short* outB,
                        float* scat, const int* scIdx,
                        const int* tArr, int tVal, int M, int relu,
                        hipStream_t stream) {
  wmma_gemm_kernel<MODE, K><<<dim3((M + 31) / 32), 256, 0, stream>>>(
      (const unsigned short*)A0, (const unsigned short*)A1, (const unsigned short*)A2,
      i0, i1, (const unsigned short*)Bt, bias, resid, outF, outB, scat, scIdx,
      tArr, tVal, M, relu);
}

// ---------- host orchestration ----------
extern "C" void kernel_launch(void* const* d_in, const int* in_sizes, int n_in,
                              void* d_out, int out_size, void* d_ws, size_t ws_size,
                              hipStream_t stream) {
  const float* x      = (const float*)d_in[0];
  const int*   ei     = (const int*)  d_in[1];
  const int*   ntype  = (const int*)  d_in[2];
  const int*   tpos   = (const int*)  d_in[3];
  const float* gat_Wq = (const float*)d_in[4];
  const float* gat_Wk = (const float*)d_in[5];
  const float* gat_Wv = (const float*)d_in[6];
  const float* gat_Wo = (const float*)d_in[7];
  const float* gat_bo = (const float*)d_in[8];
  const float* gat_g  = (const float*)d_in[9];
  const float* gat_b  = (const float*)d_in[10];
  const float* gcn_Wt = (const float*)d_in[11];
  const float* gcn_bt = (const float*)d_in[12];
  const float* gcn_mW1= (const float*)d_in[13];
  const float* gcn_mb1= (const float*)d_in[14];
  const float* gcn_mW2= (const float*)d_in[15];
  const float* gcn_mb2= (const float*)d_in[16];
  const float* gcn_aW1= (const float*)d_in[17];
  const float* gcn_ab1= (const float*)d_in[18];
  const float* gcn_aW2= (const float*)d_in[19];
  const float* gcn_ab2= (const float*)d_in[20];
  const float* gcn_Wo = (const float*)d_in[21];
  const float* gcn_bo = (const float*)d_in[22];
  const float* gcn_g  = (const float*)d_in[23];
  const float* gcn_b  = (const float*)d_in[24];
  const float* tmp_emb= (const float*)d_in[25];
  const float* tmp_W1 = (const float*)d_in[26];
  const float* tmp_b1 = (const float*)d_in[27];
  const float* tmp_W2 = (const float*)d_in[28];
  const float* tmp_b2 = (const float*)d_in[29];
  const float* tmp_Wo = (const float*)d_in[30];
  const float* tmp_bo = (const float*)d_in[31];
  const float* tmp_g  = (const float*)d_in[32];
  const float* tmp_b  = (const float*)d_in[33];
  const float* fus_W  = (const float*)d_in[34];
  const float* fus_b  = (const float*)d_in[35];
  const float* fus_g  = (const float*)d_in[36];
  const float* fus_be = (const float*)d_in[37];

  const int N = in_sizes[0] / DD;
  const int E = in_sizes[1] / 2;
  const int* row = ei;
  const int* col = ei + E;

  // bump allocator over workspace
  char* wsb = (char*)d_ws;
  size_t off = 0;
  auto alloc = [&](size_t bytes) -> void* {
    void* p = wsb + off;
    off += (bytes + 255) & ~(size_t)255;
    return p;
  };
  const size_t ND = (size_t)N * DD, ED = (size_t)E * DD;

  unsigned short* xb    = (unsigned short*)alloc(ND * 2);
  unsigned short* wqT   = (unsigned short*)alloc(DD * DD * 2);
  unsigned short* wkT   = (unsigned short*)alloc(DD * DD * 2);
  unsigned short* wvT   = (unsigned short*)alloc(DD * DD * 2);
  unsigned short* woT   = (unsigned short*)alloc(DD * DD * 2);
  unsigned short* wtT   = (unsigned short*)alloc((size_t)6 * DD * DD * 2);
  unsigned short* mW1T  = (unsigned short*)alloc((size_t)256 * DD * 2);
  unsigned short* mW2T  = (unsigned short*)alloc(DD * DD * 2);
  unsigned short* aW1T  = (unsigned short*)alloc(DD * DD * 2);
  unsigned short* aW2T  = (unsigned short*)alloc(DD * DD * 2);
  unsigned short* gWoT  = (unsigned short*)alloc(DD * DD * 2);
  unsigned short* tW1T  = (unsigned short*)alloc((size_t)256 * DD * 2);
  unsigned short* tW2T  = (unsigned short*)alloc(DD * DD * 2);
  unsigned short* tWoT  = (unsigned short*)alloc(DD * DD * 2);
  unsigned short* fWT   = (unsigned short*)alloc((size_t)384 * DD * 2);

  float* Qf      = (float*)alloc(ND * 4);
  float* Kf      = (float*)alloc(ND * 4);
  float* Vf      = (float*)alloc(ND * 4);
  float* scores  = (float*)alloc((size_t)E * NHEAD * 4);
  unsigned* mxu  = (unsigned*)alloc((size_t)N * NHEAD * 4);
  float* sm      = (float*)alloc((size_t)N * NHEAD * 4);
  float* aggGat  = (float*)alloc(ND * 4);
  unsigned short* aggGatB = (unsigned short*)alloc(ND * 2);
  float* preLN   = (float*)alloc(ND * 4);
  unsigned short* ogatB = (unsigned short*)alloc(ND * 2);

  float* hgcn    = (float*)alloc(ND * 4);
  unsigned short* hgcnB = (unsigned short*)alloc(ND * 2);
  unsigned short* em1b  = (unsigned short*)alloc(ED * 2);   // shared gcn/temporal
  float* aggGcn  = (float*)alloc(ND * 4);
  unsigned short* aggGcnB = (unsigned short*)alloc(ND * 2);
  unsigned short* a1b   = (unsigned short*)alloc(ND * 2);
  unsigned short* h2b   = (unsigned short*)alloc(ND * 2);
  unsigned short* ogcnB = (unsigned short*)alloc(ND * 2);

  float* htmpF   = (float*)alloc(ND * 4);
  unsigned short* htmpB = (unsigned short*)alloc(ND * 2);
  float* aggTmp  = (float*)alloc(ND * 4);
  unsigned short* h2tB  = (unsigned short*)alloc(ND * 2);
  unsigned short* otmpB = (unsigned short*)alloc(ND * 2);
  (void)ws_size; (void)n_in; (void)out_size;

  auto G1 = [](size_t n) { return dim3((unsigned)((n + 255) / 256)); };

  // ---- convert inputs/weights to bf16 (transposed weights) ----
  to_bf16_kernel<<<G1(ND), 256, 0, stream>>>(x, xb, (int)ND);
  conv_transpose_kernel<<<G1(DD * DD), 256, 0, stream>>>(gat_Wq, wqT, DD);
  conv_transpose_kernel<<<G1(DD * DD), 256, 0, stream>>>(gat_Wk, wkT, DD);
  conv_transpose_kernel<<<G1(DD * DD), 256, 0, stream>>>(gat_Wv, wvT, DD);
  conv_transpose_kernel<<<G1(DD * DD), 256, 0, stream>>>(gat_Wo, woT, DD);
  for (int t = 0; t < 6; ++t)
    conv_transpose_kernel<<<G1(DD * DD), 256, 0, stream>>>(gcn_Wt + (size_t)t * DD * DD,
                                                           wtT + (size_t)t * DD * DD, DD);
  conv_transpose_kernel<<<G1(256 * DD), 256, 0, stream>>>(gcn_mW1, mW1T, 256);
  conv_transpose_kernel<<<G1(DD * DD), 256, 0, stream>>>(gcn_mW2, mW2T, DD);
  conv_transpose_kernel<<<G1(DD * DD), 256, 0, stream>>>(gcn_aW1, aW1T, DD);
  conv_transpose_kernel<<<G1(DD * DD), 256, 0, stream>>>(gcn_aW2, aW2T, DD);
  conv_transpose_kernel<<<G1(DD * DD), 256, 0, stream>>>(gcn_Wo, gWoT, DD);
  conv_transpose_kernel<<<G1(256 * DD), 256, 0, stream>>>(tmp_W1, tW1T, 256);
  conv_transpose_kernel<<<G1(DD * DD), 256, 0, stream>>>(tmp_W2, tW2T, DD);
  conv_transpose_kernel<<<G1(DD * DD), 256, 0, stream>>>(tmp_Wo, tWoT, DD);
  conv_transpose_kernel<<<G1(384 * DD), 256, 0, stream>>>(fus_W, fWT, 384);

  // ---- GAT branch ----
  launch_gemm<0, DD>(xb, 0, 0, 0, 0, wqT, nullptr, nullptr, Qf, nullptr,
                     nullptr, nullptr, nullptr, 0, N, 0, stream);
  launch_gemm<0, DD>(xb, 0, 0, 0, 0, wkT, nullptr, nullptr, Kf, nullptr,
                     nullptr, nullptr, nullptr, 0, N, 0, stream);
  launch_gemm<0, DD>(xb, 0, 0, 0, 0, wvT, nullptr, nullptr, Vf, nullptr,
                     nullptr, nullptr, nullptr, 0, N, 0, stream);
  (void)hipMemsetAsync(mxu, 0, (size_t)N * NHEAD * 4, stream);
  (void)hipMemsetAsync(sm,  0, (size_t)N * NHEAD * 4, stream);
  (void)hipMemsetAsync(aggGat, 0, ND * 4, stream);
  gat_scores_kernel<<<G1(E), 256, 0, stream>>>(Qf, Kf, row, col, scores, mxu, E);
  gat_exp_kernel<<<G1((size_t)E * NHEAD), 256, 0, stream>>>(col, scores, mxu, sm, E);
  gat_agg_kernel<<<G1(ED), 256, 0, stream>>>(row, col, scores, sm, Vf, aggGat, E);
  to_bf16_kernel<<<G1(ND), 256, 0, stream>>>(aggGat, aggGatB, (int)ND);
  launch_gemm<0, DD>(aggGatB, 0, 0, 0, 0, woT, gat_bo, x, preLN, nullptr,
                     nullptr, nullptr, nullptr, 0, N, 0, stream);
  layernorm_kernel<<<G1((size_t)N * 32), 256, 0, stream>>>(preLN, gat_g, gat_b, nullptr, ogatB, N);

  // ---- Schema-aware GCN branch ----
  for (int t = 0; t < 6; ++t)   // typed transform: rows filtered by node type
    launch_gemm<0, DD>(xb, 0, 0, 0, 0, wtT + (size_t)t * DD * DD,
                       gcn_bt + (size_t)t * DD, nullptr, hgcn, hgcnB,
                       nullptr, nullptr, ntype, t, N, 0, stream);
  launch_gemm<1, 256>(hgcnB, 0, 0, row, col, mW1T, gcn_mb1, nullptr, nullptr, em1b,
                      nullptr, nullptr, nullptr, 0, E, 1, stream);   // edge MLP layer 1
  (void)hipMemsetAsync(aggGcn, 0, ND * 4, stream);
  launch_gemm<0, DD>(em1b, 0, 0, 0, 0, mW2T, gcn_mb2, nullptr, nullptr, nullptr,
                     aggGcn, col, nullptr, 0, E, 0, stream);         // layer 2 + segment sum
  to_bf16_kernel<<<G1(ND), 256, 0, stream>>>(aggGcn, aggGcnB, (int)ND);
  launch_gemm<0, DD>(aggGcnB, 0, 0, 0, 0, aW1T, gcn_ab1, nullptr, nullptr, a1b,
                     nullptr, nullptr, nullptr, 0, N, 1, stream);
  launch_gemm<0, DD>(a1b, 0, 0, 0, 0, aW2T, gcn_ab2, hgcn, nullptr, h2b,
                     nullptr, nullptr, nullptr, 0, N, 0, stream);
  launch_gemm<0, DD>(h2b, 0, 0, 0, 0, gWoT, gcn_bo, x, preLN, nullptr,
                     nullptr, nullptr, nullptr, 0, N, 0, stream);
  layernorm_kernel<<<G1((size_t)N * 32), 256, 0, stream>>>(preLN, gcn_g, gcn_b, nullptr, ogcnB, N);

  // ---- Temporal branch ----
  temporal_embed_kernel<<<G1(ND), 256, 0, stream>>>(x, tmp_emb, tpos, htmpF, htmpB, (int)ND);
  launch_gemm<1, 256>(htmpB, 0, 0, row, col, tW1T, tmp_b1, nullptr, nullptr, em1b,
                      nullptr, nullptr, nullptr, 0, E, 1, stream);
  (void)hipMemsetAsync(aggTmp, 0, ND * 4, stream);
  launch_gemm<0, DD>(em1b, 0, 0, 0, 0, tW2T, tmp_b2, nullptr, nullptr, nullptr,
                     aggTmp, col, nullptr, 0, E, 0, stream);
  add_bf16_kernel<<<G1(ND), 256, 0, stream>>>(htmpF, aggTmp, h2tB, (int)ND);
  launch_gemm<0, DD>(h2tB, 0, 0, 0, 0, tWoT, tmp_bo, nullptr, preLN, nullptr,
                     nullptr, nullptr, nullptr, 0, N, 0, stream);
  layernorm_kernel<<<G1((size_t)N * 32), 256, 0, stream>>>(preLN, tmp_g, tmp_b, nullptr, otmpB, N);

  // ---- Fusion ----
  launch_gemm<2, 384>(ogatB, ogcnB, otmpB, 0, 0, fWT, fus_b, nullptr, preLN, nullptr,
                      nullptr, nullptr, nullptr, 0, N, 0, stream);
  layernorm_kernel<<<G1((size_t)N * 32), 256, 0, stream>>>(preLN, fus_g, fus_be,
                                                           (float*)d_out, nullptr, N);
}
